// LSM_75411035783551
// MI455X (gfx1250) — matrix operand order repaired and checked by reading, not compiled, not run on hardware
//
#include <hip/hip_runtime.h>

typedef __attribute__((ext_vector_type(16))) _Float16 v16h;
typedef __attribute__((ext_vector_type(8)))  _Float16 v8h;
typedef __attribute__((ext_vector_type(8)))  float    v8f;

#define ALPHA 0.9f
#define BETA  0.85f
#define TH    1.0f

#define BB   128      // batch
#define TT   128      // time steps
#define INSZ 1024     // input size (32*32)
#define NN   2048     // reservoir neurons
#define OUTS 10       // readout size
#define KK   3072     // fused K = INSZ + NN

// ---------------- workspace layout (bytes) ----------------
static const size_t OFF_WHI  = 0;                                   // [NN][KK] f16
static const size_t OFF_WLO  = OFF_WHI + (size_t)NN * KK * 2;       // [NN][KK] f16
static const size_t OFF_A0   = OFF_WLO + (size_t)NN * KK * 2;       // [BB][KK] f16
static const size_t OFF_A1   = OFF_A0  + (size_t)BB * KK * 2;       // [BB][KK] f16
static const size_t OFF_SYN  = OFF_A1  + (size_t)BB * KK * 2;       // [BB][NN] f32
static const size_t OFF_MEM  = OFF_SYN + (size_t)BB * NN * 4;       // [BB][NN] f32
static const size_t OFF_SRO  = OFF_MEM + (size_t)BB * NN * 4;       // [BB][OUTS] f32
static const size_t OFF_MRO  = OFF_SRO + (size_t)BB * OUTS * 4;     // [BB][OUTS] f32
static const size_t OFF_BSUM = OFF_MRO + (size_t)BB * OUTS * 4;     // [NN] f32
static const size_t N_STATE_FLOATS = (size_t)2 * BB * NN + (size_t)2 * BB * OUTS;

__device__ __forceinline__ v16h load_frag16(const _Float16* p0, const _Float16* p1) {
  v8h a = *(const v8h*)p0;
  v8h b = *(const v8h*)p1;
  return __builtin_shufflevector(a, b, 0,1,2,3,4,5,6,7,8,9,10,11,12,13,14,15);
}

// ---------------- prep: split weights into f16 hi/lo, fuse biases ----------------
__global__ void lsm_prep_weights(const float* __restrict__ Win,
                                 const float* __restrict__ Wlsm,
                                 const float* __restrict__ b1,
                                 const float* __restrict__ brec,
                                 _Float16* __restrict__ Whi,
                                 _Float16* __restrict__ Wlo,
                                 float* __restrict__ bsum)
{
  const size_t stride = (size_t)gridDim.x * blockDim.x;
  const size_t gid = (size_t)blockIdx.x * blockDim.x + threadIdx.x;
  const size_t total = (size_t)NN * KK;
  for (size_t i = gid; i < total; i += stride) {
    int n = (int)(i / KK);
    int k = (int)(i % KK);
    float w = (k < INSZ) ? Win[(size_t)n * INSZ + k]
                         : Wlsm[(size_t)n * NN + (k - INSZ)];
    _Float16 hi = (_Float16)w;
    Whi[i] = hi;
    Wlo[i] = (_Float16)(w - (float)hi);
  }
  if (gid < NN) bsum[gid] = b1[gid] + brec[gid];
}

// ---------------- init: zero state, seed A-buffer0 with x_0 and zero spikes ----------------
__global__ void lsm_init(const float* __restrict__ x,
                         _Float16* __restrict__ A0,
                         float* __restrict__ stateZero)
{
  const size_t stride = (size_t)gridDim.x * blockDim.x;
  const size_t gid = (size_t)blockIdx.x * blockDim.x + threadIdx.x;
  const size_t totalA = (size_t)BB * KK;
  for (size_t i = gid; i < totalA; i += stride) {
    int b = (int)(i / KK);
    int c = (int)(i % KK);
    A0[i] = (c < INSZ) ? (_Float16)x[((size_t)b * TT + 0) * INSZ + c] : (_Float16)0.0f;
  }
  for (size_t i = gid; i < N_STATE_FLOATS; i += stride) stateZero[i] = 0.0f;
}

// ---------------- per-step fused GEMM + neuron update ----------------
// grid = (NN/64, BB/32), block = 256 (8 waves); each wave owns one 16x16 tile.
// C[b,n] = sum_k A[b,k] * W[n,k],  A = [x_t | spk_{t-1}] (f16), W split hi/lo.
__global__ __launch_bounds__(256)
void lsm_step_gemm(const _Float16* __restrict__ Aread,
                   _Float16* __restrict__ Awrite,
                   const _Float16* __restrict__ Whi,
                   const _Float16* __restrict__ Wlo,
                   float* __restrict__ syn,
                   float* __restrict__ mem,
                   const float* __restrict__ bsum)
{
  const int tid  = threadIdx.x;
  const int wave = tid >> 5;
  const int lane = tid & 31;
  const int l15  = lane & 15;
  const int hs   = lane >> 4;                 // half-select (0/1)

  const int nBase = blockIdx.x * 64 + (wave & 3) * 16;
  const int mBase = blockIdx.y * 32 + (wave >> 2) * 16;

  // A fragment: row m = mBase + l15; 16 consecutive K at kb + hs*16
  const _Float16* arow = Aread + (size_t)(mBase + l15) * KK;
  const int aoff = hs * 16;
  // B fragment: column n = nBase + l15; 16 consecutive K at kb + hs*16 (row-major [n][k])
  const _Float16* bh = Whi + (size_t)(nBase + l15) * KK + hs * 16;
  const _Float16* bl = Wlo + (size_t)(nBase + l15) * KK + hs * 16;

  v8f acc = {};
  for (int kb = 0; kb < KK; kb += 32) {
    v16h a  = load_frag16(arow + kb + aoff, arow + kb + aoff + 8);
    v16h wH = load_frag16(bh + kb,          bh + kb + 8);
    v16h wL = load_frag16(bl + kb,          bl + kb + 8);
    __builtin_prefetch(bh + kb + 256, 0, 1);   // global_prefetch_b8: stream next weights
    __builtin_prefetch(bl + kb + 256, 0, 1);
    acc = __builtin_amdgcn_wmma_f32_16x16x32_f16(false, a, false, wH, (short)0, acc, false, false);
    acc = __builtin_amdgcn_wmma_f32_16x16x32_f16(false, a, false, wL, (short)0, acc, false, false);
  }

  // Fused RSynaptic neuron update. C/D layout: lane -> column n; VGPR r -> row r + 8*hs.
  const float* accf = (const float*)&acc;
  const int n = nBase + l15;
  const float bs = bsum[n];
  #pragma unroll
  for (int r = 0; r < 8; ++r) {
    const int m = mBase + r + hs * 8;
    const size_t idx = (size_t)m * NN + n;
    float s = syn[idx];
    float u = mem[idx];
    const float reset = (u > TH) ? TH : 0.0f;   // reset from PREVIOUS mem
    s = ALPHA * s + accf[r] + bs;               // bsum = b1 + b_rec
    u = BETA * u + s - reset;
    const float spk = ((u - TH) > 0.0f) ? 1.0f : 0.0f;
    syn[idx] = s;
    mem[idx] = u;
    Awrite[(size_t)m * KK + INSZ + n] = (_Float16)spk;  // spikes for next step
  }
}

// ---------------- per-step readout + x_{t+1} conversion ----------------
// grid = BB blocks, 256 threads. Reads spikes of this step from Acur, writes out[t],
// and converts x[:, t+1, :] into the x-region of the same (next-read) buffer.
__global__ __launch_bounds__(256)
void lsm_step_readout(const _Float16* __restrict__ Acur,
                      _Float16* __restrict__ Anext,
                      const float* __restrict__ x,
                      const float* __restrict__ Wro,
                      const float* __restrict__ bro,
                      float* __restrict__ syn_ro,
                      float* __restrict__ mem_ro,
                      float* __restrict__ out,
                      int t)
{
  __shared__ float red[OUTS * 256];
  const int b   = blockIdx.x;
  const int tid = threadIdx.x;

  float partial[OUTS];
  #pragma unroll
  for (int o = 0; o < OUTS; ++o) partial[o] = 0.0f;

  const _Float16* spk = Acur + (size_t)b * KK + INSZ;
  for (int k = tid; k < NN; k += 256) {
    if ((float)spk[k] != 0.0f) {   // spikes are exactly 0/1 -> gather weight rows
      #pragma unroll
      for (int o = 0; o < OUTS; ++o) partial[o] += Wro[(size_t)o * NN + k];
    }
  }
  #pragma unroll
  for (int o = 0; o < OUTS; ++o) red[o * 256 + tid] = partial[o];
  __syncthreads();
  for (int sft = 128; sft > 0; sft >>= 1) {
    if (tid < sft) {
      #pragma unroll
      for (int o = 0; o < OUTS; ++o) red[o * 256 + tid] += red[o * 256 + tid + sft];
    }
    __syncthreads();
  }

  if (tid < OUTS) {
    const int o = tid;
    const size_t idx = (size_t)b * OUTS + o;
    const float curr = red[o * 256] + bro[o];
    float s = syn_ro[idx];
    float u = mem_ro[idx];
    const float reset = (u > TH) ? TH : 0.0f;   // TH_RO == 1.0
    s = ALPHA * s + curr;
    u = BETA * u + s - reset;
    out[(size_t)t * (BB * OUTS) + idx] = ((u - TH) > 0.0f) ? 1.0f : 0.0f;
    syn_ro[idx] = s;
    mem_ro[idx] = u;
  }

  if (t + 1 < TT) {
    const float* xrow = x + ((size_t)b * TT + (t + 1)) * INSZ;
    _Float16* arow = Anext + (size_t)b * KK;
    for (int i = tid; i < INSZ; i += 256) arow[i] = (_Float16)xrow[i];
  }
}

// ---------------- host ----------------
extern "C" void kernel_launch(void* const* d_in, const int* in_sizes, int n_in,
                              void* d_out, int out_size, void* d_ws, size_t ws_size,
                              hipStream_t stream) {
  const float* x    = (const float*)d_in[0];   // [B,T,32,32]
  const float* Win  = (const float*)d_in[1];   // [N, in]
  const float* b1   = (const float*)d_in[2];   // [N]
  const float* Wlsm = (const float*)d_in[3];   // [N, N]
  const float* brec = (const float*)d_in[4];   // [N]
  const float* Wro  = (const float*)d_in[5];   // [out, N]
  const float* bro  = (const float*)d_in[6];   // [out]
  float* out = (float*)d_out;                  // [T,B,out]

  char* ws = (char*)d_ws;
  _Float16* Whi  = (_Float16*)(ws + OFF_WHI);
  _Float16* Wlo  = (_Float16*)(ws + OFF_WLO);
  _Float16* A0   = (_Float16*)(ws + OFF_A0);
  _Float16* A1   = (_Float16*)(ws + OFF_A1);
  float*    syn  = (float*)(ws + OFF_SYN);
  float*    mem  = (float*)(ws + OFF_MEM);
  float*    sro  = (float*)(ws + OFF_SRO);
  float*    mro  = (float*)(ws + OFF_MRO);
  float*    bsum = (float*)(ws + OFF_BSUM);

  lsm_init<<<512, 256, 0, stream>>>(x, A0, syn);
  lsm_prep_weights<<<4096, 256, 0, stream>>>(Win, Wlsm, b1, brec, Whi, Wlo, bsum);

  for (int t = 0; t < TT; ++t) {
    _Float16* Ar = (t & 1) ? A1 : A0;   // holds x_t and spk_{t-1}
    _Float16* Aw = (t & 1) ? A0 : A1;   // receives spk_t (and x_{t+1} from readout)
    lsm_step_gemm<<<dim3(NN / 64, BB / 32), 256, 0, stream>>>(Ar, Aw, Whi, Wlo, syn, mem, bsum);
    lsm_step_readout<<<BB, 256, 0, stream>>>(Aw, Aw, x, Wro, bro, sro, mro, out, t);
  }
}
